// SimpleSSMBlock_23957327577495
// MI455X (gfx1250) — compile-verified, matrix operand order checked by reading, not run on hardware
//
#include <hip/hip_runtime.h>
#include <cstddef>
#include <cstdint>

// ---------------------------------------------------------------------------
// Problem constants (from reference)
// ---------------------------------------------------------------------------
#define D_MODEL 1024
#define D_STATE 16
#define D_CONV  4
#define D_INNER 2048
#define BATCH   4
#define SEQ     2048
#define ROWS    (BATCH * SEQ)        // 8192 token rows
#define LN_EPS  1e-5f

// ---------------------------------------------------------------------------
// Types for WMMA (CDNA5 wave32, v_wmma_f32_16x16x32_bf16)
// ---------------------------------------------------------------------------
typedef __attribute__((ext_vector_type(16))) __bf16 v16bf;
typedef __attribute__((ext_vector_type(8)))  float  v8f;

struct __attribute__((aligned(16))) U128 { unsigned int x[4]; };
struct __attribute__((aligned(16))) U256 { U128 lo, hi; };

static __device__ inline v16bf make_frag(U128 lo, U128 hi) {
    U256 t; t.lo = lo; t.hi = hi;
    return __builtin_bit_cast(v16bf, t);
}

static __device__ inline unsigned short f2bf(float f) {
    unsigned int u = __builtin_bit_cast(unsigned int, f);
    unsigned int r = u + 0x7FFFu + ((u >> 16) & 1u);   // round-to-nearest-even
    return (unsigned short)(r >> 16);
}

// ---------------------------------------------------------------------------
// CDNA5 async global->LDS copy path (ASYNCcnt-tracked, no VGPR staging)
// Builtin expects pointers to 16-byte int vectors (per hipcc diagnostic):
//   param0: v4i __attribute__((address_space(1)))*  (global source)
//   param1: v4i __attribute__((address_space(3)))*  (LDS destination)
// ---------------------------------------------------------------------------
#if defined(__has_builtin)
#if __has_builtin(__builtin_amdgcn_global_load_async_to_lds_b128)
#define USE_ASYNC_LDS 1
#endif
#endif

#ifdef USE_ASYNC_LDS
typedef int vint4 __attribute__((vector_size(16)));
typedef __attribute__((address_space(1))) vint4 gvint4;
typedef __attribute__((address_space(3))) vint4 lvint4;
static __device__ inline void async_b128(const unsigned short* g, unsigned short* l) {
    __builtin_amdgcn_global_load_async_to_lds_b128(
        (gvint4*)g, (lvint4*)l, 0, 0);
}
#endif

static __device__ inline void wait_async_le4() {
#ifdef USE_ASYNC_LDS
#if __has_builtin(__builtin_amdgcn_s_wait_asynccnt)
    __builtin_amdgcn_s_wait_asynccnt(4);
#else
    asm volatile("s_wait_asynccnt 0x4" ::: "memory");
#endif
#endif
}
static __device__ inline void wait_async_zero() {
#ifdef USE_ASYNC_LDS
#if __has_builtin(__builtin_amdgcn_s_wait_asynccnt)
    __builtin_amdgcn_s_wait_asynccnt(0);
#else
    asm volatile("s_wait_asynccnt 0x0" ::: "memory");
#endif
#endif
}

// ---------------------------------------------------------------------------
// fp32 -> bf16 conversion (weights + activations feeding WMMA)
// ---------------------------------------------------------------------------
__global__ __launch_bounds__(256) void f32_to_bf16_kernel(
    const float* __restrict__ in, unsigned short* __restrict__ out, int n) {
    int i = blockIdx.x * 256 + threadIdx.x;
    if (i < n) out[i] = f2bf(in[i]);
}

// ---------------------------------------------------------------------------
// WMMA GEMM: C[M,N] = A[M,K] @ W[N,K]^T   (A,W bf16 row-major; C fp32)
// EPI 0: plain store      EPI 1: softplus(C + bias[col])
// Block = 256 threads (8 wave32s), tile 128x128, K-step 32, double-buffered
// LDS fed by global_load_async_to_lds_b128 (ASYNCcnt). Each wave owns a
// 64x32 tile = 4x2 v_wmma_f32_16x16x32_bf16 accumulators.
// OOB rows are CLAMPED (not predicated): output (m,n) depends only on A-row m
// and B-row n, so stale tail rows never contaminate stored results, EXEC stays
// all-ones (WMMA requirement), and per-wave ASYNCcnt stays deterministic.
// ---------------------------------------------------------------------------
template <int EPI>
__global__ __launch_bounds__(256) void gemm_bf16_wmma(
    const unsigned short* __restrict__ A,
    const unsigned short* __restrict__ W,
    const float* __restrict__ bias,
    float* __restrict__ C,
    int M, int N, int K) {
    constexpr int BM = 128, BN = 128, BK = 32, SK = 40;   // SK: padded LDS stride
    __shared__ unsigned short As[2][BM * SK];
    __shared__ unsigned short Bs[2][BN * SK];

    const int tid  = threadIdx.x;
    const int lane = tid & 31;
    const int wave = tid >> 5;
    const int wm   = wave >> 2;     // 0..1 -> 64-row band
    const int wn   = wave & 3;      // 0..3 -> 32-col band
    const int half = lane >> 4;     // lane group (K-pair selector per ISA layout)
    const int l16  = lane & 15;
    const int m0   = blockIdx.y * BM;
    const int n0   = blockIdx.x * BN;

    // Per-thread tile-load coordinates (2x 16B chunks of A and of W per tile)
    int rowc[2], kcc[2], gmc[2], gnc[2];
#pragma unroll
    for (int c = 0; c < 2; ++c) {
        int cid = tid + c * 256;
        rowc[c] = cid >> 2;
        kcc[c]  = cid & 3;
        int gm = m0 + rowc[c]; if (gm > M - 1) gm = M - 1;   // clamp, not predicate
        int gn = n0 + rowc[c]; if (gn > N - 1) gn = N - 1;
        gmc[c] = gm; gnc[c] = gn;
    }

    auto issue_tile = [&](int k0, int buf) {
#pragma unroll
        for (int c = 0; c < 2; ++c) {
            const unsigned short* ga = A + (size_t)gmc[c] * K + k0 + kcc[c] * 8;
            const unsigned short* gw = W + (size_t)gnc[c] * K + k0 + kcc[c] * 8;
            unsigned short* la = &As[buf][rowc[c] * SK + kcc[c] * 8];
            unsigned short* lw = &Bs[buf][rowc[c] * SK + kcc[c] * 8];
#ifdef USE_ASYNC_LDS
            async_b128(ga, la);
            async_b128(gw, lw);
#else
            *(U128*)la = *(const U128*)ga;
            *(U128*)lw = *(const U128*)gw;
#endif
        }
    };

    v8f acc[4][2];
    const v8f vzero = {0.f, 0.f, 0.f, 0.f, 0.f, 0.f, 0.f, 0.f};
#pragma unroll
    for (int i = 0; i < 4; ++i)
#pragma unroll
        for (int j = 0; j < 2; ++j) acc[i][j] = vzero;

    const int kTiles = K / BK;      // K is always a multiple of 32 here
    int p = 0;
    issue_tile(0, 0);               // prologue: tile 0 -> buffer 0

    for (int kt = 0; kt < kTiles; ++kt) {
        const bool hasNext = (kt + 1 < kTiles);
        if (hasNext) issue_tile((kt + 1) * BK, p ^ 1);   // overlap next copy
        if (hasNext) wait_async_le4(); else wait_async_zero();
        __syncthreads();                                  // tile p visible to all

        // build fragments from LDS (2x ds_load_b128 each, ISA 16-bit A layout)
        v16bf afrag[4], bfrag[2];
#pragma unroll
        for (int mt = 0; mt < 4; ++mt) {
            int r = wm * 64 + mt * 16 + l16;
            U128 lo = *(const U128*)(&As[p][r * SK + half * 8]);
            U128 hi = *(const U128*)(&As[p][r * SK + 16 + half * 8]);
            afrag[mt] = make_frag(lo, hi);
        }
#pragma unroll
        for (int nt = 0; nt < 2; ++nt) {
            int r = wn * 32 + nt * 16 + l16;
            U128 lo = *(const U128*)(&Bs[p][r * SK + half * 8]);
            U128 hi = *(const U128*)(&Bs[p][r * SK + 16 + half * 8]);
            bfrag[nt] = make_frag(lo, hi);
        }

#pragma unroll
        for (int mt = 0; mt < 4; ++mt)
#pragma unroll
            for (int nt = 0; nt < 2; ++nt)
                acc[mt][nt] = __builtin_amdgcn_wmma_f32_16x16x32_bf16(
                    false, afrag[mt], false, bfrag[nt],
                    (short)0, acc[mt][nt], false, false);

        __syncthreads();   // all waves done reading buf p before it's refilled
        p ^= 1;
    }

    // store (C/D layout: VGPR r -> row r + 8*half; col = lane%16)
#pragma unroll
    for (int mt = 0; mt < 4; ++mt)
#pragma unroll
        for (int nt = 0; nt < 2; ++nt)
#pragma unroll
            for (int r = 0; r < 8; ++r) {
                int grow = m0 + wm * 64 + mt * 16 + half * 8 + r;
                int gcol = n0 + wn * 32 + nt * 16 + l16;
                if (grow < M && gcol < N) {
                    float v = acc[mt][nt][r];
                    if (EPI == 1) {
                        v += bias[gcol];
                        v = (v > 20.f) ? v : log1pf(__expf(v));
                    }
                    C[(size_t)grow * N + gcol] = v;
                }
            }
}

// ---------------------------------------------------------------------------
// Depthwise causal conv1d (+bias, +SiLU). Writes fp32 (scan) and bf16 (GEMMs).
// ---------------------------------------------------------------------------
__global__ __launch_bounds__(256) void conv_silu_kernel(
    const float* __restrict__ xz, const float* __restrict__ cw,
    const float* __restrict__ cb, float* __restrict__ xc,
    unsigned short* __restrict__ xc_bf, int total) {
    int idx = blockIdx.x * 256 + threadIdx.x;
    if (idx >= total) return;
    int d = idx % D_INNER;
    int l = (idx / D_INNER) % SEQ;
    int b = idx / (D_INNER * SEQ);
    float s = cb[d];
#pragma unroll
    for (int j = 0; j < D_CONV; ++j) {
        int ll = l + j - (D_CONV - 1);
        if (ll >= 0)
            s += cw[d * D_CONV + j] *
                 xz[((size_t)(b * SEQ + ll)) * (2 * D_INNER) + d];
    }
    s = s / (1.f + __expf(-s));     // SiLU
    size_t off = ((size_t)(b * SEQ + l)) * D_INNER + d;
    xc[off] = s;
    xc_bf[off] = f2bf(s);
}

// ---------------------------------------------------------------------------
// Selective scan: one lane per (b,d); 16 states in registers; B/C staged in
// LDS per 16-step chunk. Fused D-skip + SiLU(z) gate; emits bf16 y for GEMM.
// ---------------------------------------------------------------------------
__global__ __launch_bounds__(256) void scan_kernel(
    const float* __restrict__ dt, const float* __restrict__ bx,
    const float* __restrict__ xc, const float* __restrict__ xz,
    const float* __restrict__ A_log, const float* __restrict__ Dvec,
    unsigned short* __restrict__ y_bf) {
    constexpr int CH = 16;
    __shared__ float sBC[CH * 32];         // [t][0:16)=B, [t][16:32)=C
    const int b = blockIdx.x >> 3;         // 8 blocks per batch
    const int d = ((blockIdx.x & 7) << 8) + threadIdx.x;

    float Areg[D_STATE], h[D_STATE];
#pragma unroll
    for (int s = 0; s < D_STATE; ++s) {
        Areg[s] = -__expf(A_log[d * D_STATE + s]);
        h[s] = 0.f;
    }
    const float Dd = Dvec[d];

    for (int l0 = 0; l0 < SEQ; l0 += CH) {
        int base = (b * SEQ + l0) * 32;    // bx rows are contiguous 32 floats
        sBC[threadIdx.x]       = bx[base + threadIdx.x];
        sBC[threadIdx.x + 256] = bx[base + threadIdx.x + 256];
        __syncthreads();
#pragma unroll 4
        for (int t = 0; t < CH; ++t) {
            int l = l0 + t;
            size_t off = ((size_t)(b * SEQ + l)) * D_INNER + d;
            float dtv = dt[off];
            float xv  = xc[off];
            float zv  = xz[((size_t)(b * SEQ + l)) * (2 * D_INNER) + D_INNER + d];
            float y = 0.f;
#pragma unroll
            for (int s = 0; s < D_STATE; ++s) {
                float dA = __expf(Areg[s] * dtv);
                h[s] = dA * h[s] + (dtv * sBC[t * 32 + s]) * xv;
                y += h[s] * sBC[t * 32 + 16 + s];
            }
            float yo = y + xv * Dd;
            float g  = zv / (1.f + __expf(-zv));
            y_bf[off] = f2bf(yo * g);
        }
        __syncthreads();
    }
}

// ---------------------------------------------------------------------------
// Residual add + LayerNorm over D_MODEL=1024 (wave32 shuffle reduction)
// ---------------------------------------------------------------------------
__global__ __launch_bounds__(256) void residual_ln_kernel(
    const float* __restrict__ out_pre, const float* __restrict__ x,
    const float* __restrict__ g, const float* __restrict__ beta,
    float* __restrict__ out) {
    const int row = blockIdx.x;
    const float* pr = out_pre + (size_t)row * D_MODEL;
    const float* xr = x + (size_t)row * D_MODEL;

    float vals[4], s = 0.f, ss = 0.f;
#pragma unroll
    for (int i = 0; i < 4; ++i) {
        int c = threadIdx.x + i * 256;
        float v = pr[c] + xr[c];
        vals[i] = v;
        s += v; ss += v * v;
    }
#pragma unroll
    for (int off = 16; off > 0; off >>= 1) {   // wave32 reduction
        s  += __shfl_down(s,  off, 32);
        ss += __shfl_down(ss, off, 32);
    }
    __shared__ float rs[8], rss[8];
    int wave = threadIdx.x >> 5, lane = threadIdx.x & 31;
    if (lane == 0) { rs[wave] = s; rss[wave] = ss; }
    __syncthreads();
    if (threadIdx.x == 0) {
        float a = 0.f, b2 = 0.f;
        for (int w = 0; w < 8; ++w) { a += rs[w]; b2 += rss[w]; }
        rs[0] = a; rss[0] = b2;
    }
    __syncthreads();
    float mean = rs[0] * (1.f / D_MODEL);
    float var  = rss[0] * (1.f / D_MODEL) - mean * mean;
    float rstd = rsqrtf(var + LN_EPS);
#pragma unroll
    for (int i = 0; i < 4; ++i) {
        int c = threadIdx.x + i * 256;
        out[(size_t)row * D_MODEL + c] = (vals[i] - mean) * rstd * g[c] + beta[c];
    }
}

// ---------------------------------------------------------------------------
// Host-side orchestration
// ---------------------------------------------------------------------------
extern "C" void kernel_launch(void* const* d_in, const int* in_sizes, int n_in,
                              void* d_out, int out_size, void* d_ws, size_t ws_size,
                              hipStream_t stream) {
    (void)in_sizes; (void)n_in; (void)out_size; (void)ws_size;
    const float* x         = (const float*)d_in[0];
    const float* in_proj_w = (const float*)d_in[1];
    const float* conv_w    = (const float*)d_in[2];
    const float* conv_b    = (const float*)d_in[3];
    const float* x_proj_w  = (const float*)d_in[4];
    const float* dt_proj_w = (const float*)d_in[5];
    const float* dt_proj_b = (const float*)d_in[6];
    const float* A_log     = (const float*)d_in[7];
    const float* Dvec      = (const float*)d_in[8];
    const float* out_proj_w= (const float*)d_in[9];
    const float* ln_g      = (const float*)d_in[10];
    const float* ln_b      = (const float*)d_in[11];
    float* out = (float*)d_out;

    // workspace partition (256-byte aligned)
    char* w = (char*)d_ws;
    size_t off = 0;
    auto alloc = [&](size_t bytes) -> void* {
        void* p = w + off;
        off = (off + bytes + 255) & ~(size_t)255;
        return p;
    };
    unsigned short* x_bf     = (unsigned short*)alloc((size_t)ROWS * D_MODEL * 2);
    unsigned short* w_in_bf  = (unsigned short*)alloc((size_t)2 * D_INNER * D_MODEL * 2);
    unsigned short* w_dt_bf  = (unsigned short*)alloc((size_t)D_INNER * D_INNER * 2);
    unsigned short* w_xp_bf  = (unsigned short*)alloc((size_t)2 * D_STATE * D_INNER * 2);
    unsigned short* w_out_bf = (unsigned short*)alloc((size_t)D_MODEL * D_INNER * 2);
    float*          xz       = (float*)alloc((size_t)ROWS * 2 * D_INNER * 4);
    float*          xc       = (float*)alloc((size_t)ROWS * D_INNER * 4);
    unsigned short* xc_bf    = (unsigned short*)alloc((size_t)ROWS * D_INNER * 2);
    float*          dt       = (float*)alloc((size_t)ROWS * D_INNER * 4);
    float*          bx       = (float*)alloc((size_t)ROWS * 2 * D_STATE * 4);
    unsigned short* y_bf     = (unsigned short*)alloc((size_t)ROWS * D_INNER * 2);
    float*          out_pre  = (float*)alloc((size_t)ROWS * D_MODEL * 4);

    auto cvt = [&](const float* src, unsigned short* dst, int n) {
        f32_to_bf16_kernel<<<(n + 255) / 256, 256, 0, stream>>>(src, dst, n);
    };
    // bf16 conversions
    cvt(x,          x_bf,     ROWS * D_MODEL);
    cvt(in_proj_w,  w_in_bf,  2 * D_INNER * D_MODEL);
    cvt(dt_proj_w,  w_dt_bf,  D_INNER * D_INNER);
    cvt(x_proj_w,   w_xp_bf,  2 * D_STATE * D_INNER);
    cvt(out_proj_w, w_out_bf, D_MODEL * D_INNER);

    // 1) in_proj GEMM: (8192x1024)@(1024x4096) -> xz
    gemm_bf16_wmma<0><<<dim3((2 * D_INNER + 127) / 128, (ROWS + 127) / 128),
                        256, 0, stream>>>(x_bf, w_in_bf, nullptr, xz,
                                          ROWS, 2 * D_INNER, D_MODEL);
    // 2) depthwise conv + SiLU
    {
        int total = ROWS * D_INNER;
        conv_silu_kernel<<<(total + 255) / 256, 256, 0, stream>>>(
            xz, conv_w, conv_b, xc, xc_bf, total);
    }
    // 3) dt GEMM with fused softplus(+bias): (8192x2048)@(2048x2048)
    gemm_bf16_wmma<1><<<dim3((D_INNER + 127) / 128, (ROWS + 127) / 128),
                        256, 0, stream>>>(xc_bf, w_dt_bf, dt_proj_b, dt,
                                          ROWS, D_INNER, D_INNER);
    // 4) x_proj GEMM: (8192x2048)@(2048x32) -> B/C
    gemm_bf16_wmma<0><<<dim3(1, (ROWS + 127) / 128), 256, 0, stream>>>(
        xc_bf, w_xp_bf, nullptr, bx, ROWS, 2 * D_STATE, D_INNER);
    // 5) selective scan (+D skip, +SiLU gate) -> bf16 y
    scan_kernel<<<BATCH * (D_INNER / 256), 256, 0, stream>>>(
        dt, bx, xc, xz, A_log, Dvec, y_bf);
    // 6) out_proj GEMM: (8192x2048)@(2048x1024)
    gemm_bf16_wmma<0><<<dim3((D_MODEL + 127) / 128, (ROWS + 127) / 128),
                        256, 0, stream>>>(y_bf, w_out_bf, nullptr, out_pre,
                                          ROWS, D_MODEL, D_INNER);
    // 7) residual add + LayerNorm
    residual_ln_kernel<<<ROWS, 256, 0, stream>>>(out_pre, x, ln_g, ln_b, out);
}
